// SA_Layer2_88716844466212
// MI455X (gfx1250) — matrix operand order repaired and checked by reading, not compiled, not run on hardware
//
#include <hip/hip_runtime.h>
#include <hip/hip_bf16.h>

// ---------------------------------------------------------------------------
// Shapes (fixed by the reference): B=8, C=512, H=1024, W=8, C4=128
//   HW = H*W = 8192 pixels per (b, channel-plane)
// ---------------------------------------------------------------------------

typedef __attribute__((ext_vector_type(16))) __bf16 v16bf;
typedef __attribute__((ext_vector_type(8)))  float  v8f;

union BfFrag {
    v16bf v;
    unsigned int u[8];
};

__device__ __forceinline__ unsigned int f2bf(float f) {
    // round-to-nearest-even fp32 -> bf16 (result in low 16 bits)
    unsigned int u = __float_as_uint(f);
    u += 0x7FFFu + ((u >> 16) & 1u);
    return u >> 16;
}
__device__ __forceinline__ unsigned int pack2bf(float lo, float hi) {
    return f2bf(lo) | (f2bf(hi) << 16);
}

#define BM  128
#define BN  128
#define BK  32
#define LDP 34   // padded LDS row pitch in u16 (68 B rows: 4B-aligned, odd bank step)

// ---------------------------------------------------------------------------
// Generic strided GEMM:  C[m,n] (op)= sum_k A[m,k] * B[k,n]
//   fp32 global -> bf16 LDS -> v_wmma_f32_16x16x32_bf16 (f32 accumulate).
// Template: AK1 -> A rows contiguous in k (float4 loads); BN1 -> B rows
// contiguous in n (float4 loads, bf16 scatter into transposed LDS tile).
// Per-slice (blockIdx.z) offsets: z -> (zh=z/ZW, zl=z%ZW), base += zh*oh+zl*ol.
// mode: 0 -> C = acc ; 1 -> C = acc + bias[m] ; 2 -> C = X2[m,n] - acc
// All M,N multiples of 128, K multiple of 32: no divergence, EXEC all-ones
// around the WMMAs as the ISA requires.
// ---------------------------------------------------------------------------
template <bool AK1, bool BN1>
__global__ __launch_bounds__(256)
void wmma_gemm(const float* __restrict__ A, const float* __restrict__ B,
               float* __restrict__ C, const float* __restrict__ X2,
               const float* __restrict__ bias,
               int M, int N, int K,
               long a_sm, long a_sk, long b_sk, long b_sn,
               long c_sm, long c_sn,
               int ZW,
               long a_oh, long a_ol, long b_oh, long b_ol,
               long c_oh, long c_ol,
               int mode)
{
    __shared__ __align__(16) unsigned short As[BM * LDP];
    __shared__ __align__(16) unsigned short Bs[BN * LDP]; // transposed: Bs[n*LDP + k]

    const int tid   = threadIdx.x;
    const int wave  = tid >> 5;
    const int lane  = tid & 31;
    const int lhalf = lane >> 4;   // 0 / 1
    const int l16   = lane & 15;

    const int z  = blockIdx.z;
    const int zh = z / ZW;
    const int zl = z - zh * ZW;

    const float* Ab = A + (long)zh * a_oh + (long)zl * a_ol;
    const float* Bb = B + (long)zh * b_oh + (long)zl * b_ol;
    float*       Cb = C + (long)zh * c_oh + (long)zl * c_ol;
    const float* Xb = (X2 != nullptr) ? (X2 + (long)zh * c_oh + (long)zl * c_ol) : nullptr;

    const int m0 = blockIdx.y * BM;
    const int n0 = blockIdx.x * BN;

    const int wm = wave >> 2;  // 0..1 : which 64-row half
    const int wn = wave & 3;   // 0..3 : which 32-col quarter

    v8f acc[4][2];
#pragma unroll
    for (int mi = 0; mi < 4; ++mi)
#pragma unroll
        for (int ni = 0; ni < 2; ++ni)
            acc[mi][ni] = (v8f){0.f, 0.f, 0.f, 0.f, 0.f, 0.f, 0.f, 0.f};

    for (int k0 = 0; k0 < K; k0 += BK) {
        if (k0 + BK < K) {
            // warm caches for the next K-tile (global_prefetch_b8, near scope)
            __builtin_prefetch((const void*)(Bb + (long)(k0 + BK) * b_sk + (long)n0 * b_sn), 0, 3);
            __builtin_prefetch((const void*)(Ab + (long)m0 * a_sm + (long)(k0 + BK) * a_sk), 0, 3);
        }

        // ---- stage A tile (BM x BK) ----
        if (AK1) {
            // contiguous k: float4 (b128) global loads, packed b32 LDS stores
#pragma unroll
            for (int it = 0; it < 4; ++it) {
                int idx = tid + it * 256;       // 1024 float4-chunks
                int m  = idx >> 3;              // 8 chunks per row
                int kc = (idx & 7) * 4;
                const float4 vv = *(const float4*)(Ab + (long)(m0 + m) * a_sm + (long)(k0 + kc));
                unsigned short* dst = &As[m * LDP + kc];
                *(unsigned int*)(dst)     = pack2bf(vv.x, vv.y);
                *(unsigned int*)(dst + 2) = pack2bf(vv.z, vv.w);
            }
        } else {
            // strided k: gather pairs, packed b32 LDS stores
#pragma unroll
            for (int it = 0; it < 8; ++it) {
                int idx = tid + it * 256;       // 2048 k-pairs
                int m  = idx & 127;             // consecutive tid -> consecutive m
                int kp = (idx >> 7) * 2;
                long base = (long)(m0 + m) * a_sm + (long)(k0 + kp) * a_sk;
                float f0 = Ab[base];
                float f1 = Ab[base + a_sk];
                *(unsigned int*)&As[m * LDP + kp] = pack2bf(f0, f1);
            }
        }

        // ---- stage B tile (BK x BN), transposed into Bs[n][k] ----
        if (BN1) {
            // contiguous n: float4 (b128) global loads, bf16 scatter stores
#pragma unroll
            for (int it = 0; it < 4; ++it) {
                int idx = tid + it * 256;       // 1024 float4-chunks
                int k  = idx >> 5;              // 32 chunks per k-row
                int nc = (idx & 31) * 4;
                const float4 vv = *(const float4*)(Bb + (long)(k0 + k) * b_sk + (long)(n0 + nc));
                Bs[(nc + 0) * LDP + k] = (unsigned short)f2bf(vv.x);
                Bs[(nc + 1) * LDP + k] = (unsigned short)f2bf(vv.y);
                Bs[(nc + 2) * LDP + k] = (unsigned short)f2bf(vv.z);
                Bs[(nc + 3) * LDP + k] = (unsigned short)f2bf(vv.w);
            }
        } else {
            // strided n: gather k-pairs per n, packed b32 LDS stores
#pragma unroll
            for (int it = 0; it < 8; ++it) {
                int idx = tid + it * 256;       // 2048 k-pairs
                int n  = idx & 127;             // consecutive tid -> consecutive n
                int kp = (idx >> 7) * 2;
                long base = (long)(k0 + kp) * b_sk + (long)(n0 + n) * b_sn;
                float f0 = Bb[base];
                float f1 = Bb[base + b_sk];
                *(unsigned int*)&Bs[n * LDP + kp] = pack2bf(f0, f1);
            }
        }
        __syncthreads();

        // ---- load fragments per ISA 7.12.2 layouts ----
        BfFrag af[4];
#pragma unroll
        for (int mi = 0; mi < 4; ++mi) {
            int m = wm * 64 + mi * 16 + l16;
#pragma unroll
            for (int r = 0; r < 8; ++r) {
                int kk = (r >> 2) * 16 + lhalf * 8 + (r & 3) * 2; // A: VGPR r -> K pair
                af[mi].u[r] = *(const unsigned int*)&As[m * LDP + kk];
            }
        }
        BfFrag bfr[2];
#pragma unroll
        for (int ni = 0; ni < 2; ++ni) {
            int n = wn * 32 + ni * 16 + l16;
#pragma unroll
            for (int r = 0; r < 8; ++r) {
                int kk = (r >> 2) * 16 + lhalf * 8 + (r & 3) * 2; // B: VGPR r -> K pair
                bfr[ni].u[r] = *(const unsigned int*)&Bs[n * LDP + kk];
            }
        }

#pragma unroll
        for (int mi = 0; mi < 4; ++mi)
#pragma unroll
            for (int ni = 0; ni < 2; ++ni)
                acc[mi][ni] = __builtin_amdgcn_wmma_f32_16x16x32_bf16(
                    false, af[mi].v, false, bfr[ni].v,
                    (short)0, acc[mi][ni], false, false);

        __syncthreads();
    }

    // ---- epilogue: C/D layout -> lane = N, VGPR r = row (r + 8*lane_half) ----
#pragma unroll
    for (int mi = 0; mi < 4; ++mi) {
#pragma unroll
        for (int ni = 0; ni < 2; ++ni) {
            int ncol = n0 + wn * 32 + ni * 16 + l16;
#pragma unroll
            for (int r = 0; r < 8; ++r) {
                int mrow = m0 + wm * 64 + mi * 16 + r + 8 * lhalf;
                long cidx = (long)mrow * c_sm + (long)ncol * c_sn;
                float va = acc[mi][ni][r];
                float o;
                if (mode == 1)      o = va + bias[mrow];
                else if (mode == 2) o = Xb[cidx] - va;     // u = x - x_r
                else                o = va;
                Cb[cidx] = o;
            }
        }
    }
}

// ---------------------------------------------------------------------------
// Column softmax over i for e[slice][i][j], slice = b*8+w, in place.
// softmax sum == the normalizer, so the reference's L1 renorm reduces to a
// division by (1 + 1e-9); folded into inv.
// grid.x = 64 slices * 4 column-tiles, block = 256 (one column per thread).
// ---------------------------------------------------------------------------
__global__ __launch_bounds__(256)
void softmax_col(float* __restrict__ e)
{
    const int slice = blockIdx.x >> 2;
    const int j = ((blockIdx.x & 3) << 8) + threadIdx.x;
    float* col = e + (size_t)slice * 1048576 + j;

    float m = -3.4e38f, s = 0.f;
    for (int i = 0; i < 1024; ++i) {
        float v = col[(size_t)i * 1024];
        float nm = fmaxf(m, v);
        s = s * __expf(m - nm) + __expf(v - nm);
        m = nm;
    }
    float inv = 1.0f / (s * (1.0f + 1e-9f));
    for (int i = 0; i < 1024; ++i) {
        size_t off = (size_t)i * 1024;
        col[off] = __expf(col[off] - m) * inv;
    }
}

// ---------------------------------------------------------------------------
// BatchNorm training-mode batch stats: per channel c over (B,H,W) = 65536
// elements of t[b][c][hw].  stats[c] = mean, stats[512+c] = rsqrt(var+eps).
// ---------------------------------------------------------------------------
__global__ __launch_bounds__(256)
void bn_stats(const float* __restrict__ t, float* __restrict__ stats)
{
    const int c = blockIdx.x;
    float s = 0.f, s2 = 0.f;
    for (int b = 0; b < 8; ++b) {
        const float* p = t + ((size_t)b * 512 + c) * 8192;
        for (int i = threadIdx.x; i < 8192; i += 256) {
            float v = p[i];
            s  += v;
            s2 += v * v;
        }
    }
    __shared__ float rs[256], rq[256];
    rs[threadIdx.x] = s;
    rq[threadIdx.x] = s2;
    __syncthreads();
    for (int o = 128; o > 0; o >>= 1) {
        if (threadIdx.x < o) {
            rs[threadIdx.x] += rs[threadIdx.x + o];
            rq[threadIdx.x] += rq[threadIdx.x + o];
        }
        __syncthreads();
    }
    if (threadIdx.x == 0) {
        float mean = rs[0] * (1.0f / 65536.0f);
        float var  = rq[0] * (1.0f / 65536.0f) - mean * mean; // biased
        stats[c]       = mean;
        stats[512 + c] = rsqrtf(var + 1e-5f);
    }
}

// ---------------------------------------------------------------------------
// out = x + relu(gamma * (t - mean) * inv_std + beta), elementwise.
// ---------------------------------------------------------------------------
__global__ __launch_bounds__(256)
void bn_relu_add(const float* __restrict__ x, const float* __restrict__ t,
                 const float* __restrict__ stats,
                 const float* __restrict__ gamma, const float* __restrict__ beta,
                 float* __restrict__ out)
{
    size_t idx = (size_t)blockIdx.x * 256 + threadIdx.x;
    int c = (int)((idx >> 13) & 511);   // (idx / 8192) % 512
    float tn = gamma[c] * (t[idx] - stats[c]) * stats[512 + c] + beta[c];
    out[idx] = x[idx] + fmaxf(tn, 0.f);
}

// ---------------------------------------------------------------------------
// Orchestration
// ---------------------------------------------------------------------------
extern "C" void kernel_launch(void* const* d_in, const int* in_sizes, int n_in,
                              void* d_out, int out_size, void* d_ws, size_t ws_size,
                              hipStream_t stream)
{
    (void)in_sizes; (void)n_in; (void)out_size; (void)ws_size;

    const float* x     = (const float*)d_in[0];  // [8,512,1024,8]
    const float* qk_w  = (const float*)d_in[1];  // [128,512]
    const float* v_w   = (const float*)d_in[2];  // [512,512]
    const float* v_b   = (const float*)d_in[3];  // [512]
    const float* t_w   = (const float*)d_in[4];  // [512,512]
    const float* t_b   = (const float*)d_in[5];  // [512]
    const float* gamma = (const float*)d_in[6];  // [512]
    const float* beta  = (const float*)d_in[7];  // [512]
    float* out = (float*)d_out;

    char* ws = (char*)d_ws;
    const size_t OFF_Q = 0;                        //  32 MB : q  [8][128][8192] f32
    const size_t OFF_V = OFF_Q + 33554432ull;      // 128 MB : v  [8][512][8192] f32
    const size_t OFF_E = OFF_V + 134217728ull;     // 256 MB : e/att [64][1024][1024] f32
    const size_t OFF_U = OFF_E + 268435456ull;     // 128 MB : u = x - x_r [8][512][8192] f32
    const size_t OFF_S = OFF_U + 134217728ull;     //   4 KB : stats [2][512]
    float* q     = (float*)(ws + OFF_Q);
    float* v     = (float*)(ws + OFF_V);
    float* e     = (float*)(ws + OFF_E);
    float* u     = (float*)(ws + OFF_U);
    float* t     = e;                              // alias: e is dead once u exists
    float* stats = (float*)(ws + OFF_S);

    dim3 blk(256);

    // GEMM1: q[b] = qk_w(128x512) * x_b(512x8192)
    wmma_gemm<true, true><<<dim3(64, 1, 8), blk, 0, stream>>>(
        qk_w, x, q, nullptr, nullptr,
        128, 8192, 512,
        /*a_sm*/512, /*a_sk*/1, /*b_sk*/8192, /*b_sn*/1,
        /*c_sm*/8192, /*c_sn*/1,
        /*ZW*/1,
        /*a_oh*/0, /*a_ol*/0, /*b_oh*/4194304, /*b_ol*/0,
        /*c_oh*/1048576, /*c_ol*/0, /*mode*/0);

    // GEMM2: v[b] = v_w(512x512) * x_b + v_b
    wmma_gemm<true, true><<<dim3(64, 4, 8), blk, 0, stream>>>(
        v_w, x, v, nullptr, v_b,
        512, 8192, 512,
        512, 1, 8192, 1,
        8192, 1,
        1,
        0, 0, 4194304, 0,
        4194304, 0, /*mode*/1);

    // GEMM3: e[b,w] = q_bw^T(1024x128) * q_bw(128x1024)   (slices z = b*8+w)
    wmma_gemm<false, false><<<dim3(8, 8, 64), blk, 0, stream>>>(
        q, q, e, nullptr, nullptr,
        1024, 1024, 128,
        /*a_sm(i)*/8, /*a_sk(d)*/8192, /*b_sk(d)*/8192, /*b_sn(j)*/8,
        /*c_sm(i)*/1024, /*c_sn(j)*/1,
        /*ZW*/8,
        /*a_oh*/1048576, /*a_ol*/1, /*b_oh*/1048576, /*b_ol*/1,
        /*c_oh*/8388608, /*c_ol*/1048576, /*mode*/0);

    // softmax over i (in place), + L1 renorm fold
    softmax_col<<<dim3(256), blk, 0, stream>>>(e);

    // GEMM4: u[b,:,:,w] = x - v_bw(512x1024) * att_bw(1024x1024)
    wmma_gemm<false, true><<<dim3(8, 4, 64), blk, 0, stream>>>(
        v, e, u, x, nullptr,
        512, 1024, 1024,
        /*a_sm(c)*/8192, /*a_sk(i)*/8, /*b_sk(i)*/1024, /*b_sn(j)*/1,
        /*c_sm(c)*/8192, /*c_sn(j)*/8,
        /*ZW*/8,
        /*a_oh*/4194304, /*a_ol*/1, /*b_oh*/8388608, /*b_ol*/1048576,
        /*c_oh*/4194304, /*c_ol*/1, /*mode*/2);

    // GEMM5: t[b] = t_w(512x512) * u_b(512x8192) + t_b
    wmma_gemm<true, true><<<dim3(64, 4, 8), blk, 0, stream>>>(
        t_w, u, t, nullptr, t_b,
        512, 8192, 512,
        512, 1, 8192, 1,
        8192, 1,
        1,
        0, 0, 4194304, 0,
        4194304, 0, /*mode*/1);

    // BN batch stats per channel
    bn_stats<<<dim3(512), blk, 0, stream>>>(t, stats);

    // out = x + relu(BN(t))
    bn_relu_add<<<dim3(131072), blk, 0, stream>>>(x, t, stats, gamma, beta, out);
}
